// DeepResidualLagrangianNetwork_77841987273371
// MI455X (gfx1250) — matrix operand order repaired and checked by reading, not compiled, not run on hardware
//
#include <hip/hip_runtime.h>
#include <hip/hip_bf16.h>
#include <cstdint>

// ---------------------------------------------------------------------------
// DeLaN inverse dynamics for MI455X (gfx1250, wave32, WMMA bf16 16x16x32).
//
// Pipeline (all bf16 intermediates, f32 WMMA accumulate):
//   1. prep:      W1,W2 -> bf16; Wcat = [Wld;Wlo;Wg;0] (384x512), bcat
//   2. layer0:    y0 = relu(q@W0^T+b0)                   (scalar, K=24)
//   3. gemm h1:   relu(y0@W1^T+b1)        WMMA  4096x512x512
//   4. gemm y2:   relu(h1@W2^T+b2)        WMMA
//   5. gemm lhead: y2@Wcat^T+bcat         WMMA  4096x384x512 (pre-relu a's)
//   6. hact:      mask array (a_ld for diag rows, 1 elsewhere)
//   7. residual:  h1->8->4->1 MLP         (scalar)
//   8. der0:      der0[bn][o] = (y0[b,o]>0)*W0[o,n]
//   9. gemm der1: (der0@W1^T) masked by h1>0    WMMA  98304x512x512
//  10. gemm der2: (der1@W2^T) masked by y2>0    WMMA
//  11. gemm derH: (der2@Wcat^T) masked by hact  WMMA  98304x384x512
//  12. assemble:  per-batch wave32: L, H, Hdt, quad terms -> tau, power
//
// GEMM: 128x256 block tile, 8 waves of 64x64 (16 WMMA accums each),
// double-buffered LDS fed by GLOBAL_LOAD_ASYNC_TO_LDS_B128 (ASYNCcnt),
// 16 ds_load_b128 : 16 v_wmma per K-step (1:1).
// ---------------------------------------------------------------------------

typedef __bf16 bf16_t;
typedef __attribute__((ext_vector_type(16))) __bf16 bf16x16;
typedef __attribute__((ext_vector_type(8)))  float  floatx8;
typedef int v4i __attribute__((vector_size(16)));

#define N_DOF  24
#define WIDTH  512
#define BATCH  4096
#define M_TRIL 300
#define HM     384            // padded head-row count (3 x 128 tiles)
#define BN     (BATCH * N_DOF)
#define EPSV   1e-5f

#if __has_builtin(__builtin_amdgcn_global_load_async_to_lds_b128)
#define HAVE_ASYNC_LDS 1
#else
#define HAVE_ASYNC_LDS 0
#endif

#define AS1 __attribute__((address_space(1)))
#define AS3 __attribute__((address_space(3)))

union PackBF8 { uint4 u; bf16_t h[8]; };

__device__ __forceinline__ floatx8 wmma_bf16(bf16x16 a, bf16x16 b, floatx8 c) {
  // (neg_a, A, neg_b, B, c_mod, C, reuse_a, reuse_b)
  return __builtin_amdgcn_wmma_f32_16x16x32_bf16(false, a, false, b,
                                                 (short)0, c, false, false);
}

#if HAVE_ASYNC_LDS
__device__ __forceinline__ void async_cp16(const void* g, void* l) {
  // (global v4i* src, lds v4i* dst, imm offset, imm cpol)
  __builtin_amdgcn_global_load_async_to_lds_b128(
      (AS1 v4i*)g, (AS3 v4i*)l, 0, 0);
}
template<int N>
__device__ __forceinline__ void wait_async() {
#if __has_builtin(__builtin_amdgcn_s_wait_asynccnt)
  __builtin_amdgcn_s_wait_asynccnt((unsigned short)N);
#else
  asm volatile("s_wait_asynccnt %0" :: "i"(N) : "memory");
#endif
}
#endif

// ---------------------------------------------------------------------------
// 1. Weight prep: fp32 -> bf16, build concatenated head weight matrix.
// ---------------------------------------------------------------------------
__global__ void prep_kernel(const float* __restrict__ W1, const float* __restrict__ W2,
                            const float* __restrict__ Wld, const float* __restrict__ Wlo,
                            const float* __restrict__ Wg,
                            const float* __restrict__ bld, const float* __restrict__ blo,
                            const float* __restrict__ bg,
                            bf16_t* __restrict__ W1b, bf16_t* __restrict__ W2b,
                            bf16_t* __restrict__ Wcat, float* __restrict__ bcat)
{
  int idx = blockIdx.x * 256 + threadIdx.x;
  const int n1 = WIDTH * WIDTH;
  if (idx < n1) { W1b[idx] = (bf16_t)W1[idx]; return; }
  idx -= n1;
  if (idx < n1) { W2b[idx] = (bf16_t)W2[idx]; return; }
  idx -= n1;
  const int nc = HM * WIDTH;
  if (idx < nc) {
    int row = idx / WIDTH, col = idx % WIDTH;
    float v = 0.f;
    if (row < N_DOF)        v = Wld[row * WIDTH + col];
    else if (row < M_TRIL)  v = Wlo[(row - N_DOF) * WIDTH + col];
    else if (row == M_TRIL) v = Wg[col];
    Wcat[row * WIDTH + col] = (bf16_t)v;
    return;
  }
  idx -= nc;
  if (idx < HM) {
    float v = 0.f;
    if (idx < N_DOF)        v = bld[idx];
    else if (idx < M_TRIL)  v = blo[idx - N_DOF];
    else if (idx == M_TRIL) v = bg[0];
    bcat[idx] = v;
  }
}

// ---------------------------------------------------------------------------
// 2. Layer 0 (K = 24, scalar is fine).
// ---------------------------------------------------------------------------
__global__ void layer0_kernel(const float* __restrict__ q, const float* __restrict__ W0,
                              const float* __restrict__ b0, bf16_t* __restrict__ y0b)
{
  int idx = blockIdx.x * 256 + threadIdx.x;  // b*512 + o, exact grid
  int b = idx >> 9, o = idx & 511;
  float a = b0[o];
  const float* qr = q + b * N_DOF;
  const float* wr = W0 + o * N_DOF;
#pragma unroll
  for (int i = 0; i < N_DOF; ++i) a += qr[i] * wr[i];
  y0b[idx] = (bf16_t)fmaxf(a, 0.f);
}

// ---------------------------------------------------------------------------
// WMMA GEMM:  Out[n][o] = epilogue( sum_k W[o][k] * D[n][k] )
//   W: [M][K] bf16 row-major (M = gridDim.y*128, K mult of 32)
//   D: [N][K] bf16 row-major (N = gridDim.x*256)
// Block 256 threads = 8 waves in 2(M) x 4(N); wave tile 64x64 = 4x4 WMMAs.
// LDS stride 80 B (32 bf16 + 16 B pad): 16 B alignment for ds_b128 and bank
// spread. Double-buffered: async DMA of K-tile s+1 overlaps WMMA on tile s;
// s_wait_asynccnt<=6 drains exactly the previous tile (in-order completion).
// ---------------------------------------------------------------------------
template<bool BIAS, bool RELU, bool MASK>
__global__ __launch_bounds__(256)
void gemm_bf16_wmma(const bf16_t* __restrict__ W, const bf16_t* __restrict__ D,
                    bf16_t* __restrict__ Out, const float* __restrict__ bias,
                    const bf16_t* __restrict__ mask,
                    int K, int nPerB, int maskStride, int outStride)
{
  constexpr int LSTR   = 80;                     // bytes per 32-bf16 LDS row
  constexpr int ABYTES = 128 * LSTR;             // A tile (128 rows)
  constexpr int BUFB   = (128 + 256) * LSTR;     // one double-buffer half
  __shared__ uint4 lds4[2 * BUFB / 16];          // 60 KB
  char* lds = (char*)lds4;

  const int nBlk = blockIdx.x * 256;
  const int mBlk = blockIdx.y * 128;
  const int tid  = threadIdx.x;
  const int wid  = tid >> 5;
  const int lane = tid & 31;
  const int wm   = wid >> 2;               // 0..1 -> M offset wm*64
  const int wn   = wid & 3;                // 0..3 -> N offset wn*64
  const int l15  = lane & 15;
  const int hi   = lane >> 4;              // lane half selects K sub-range

  floatx8 acc[4][4] = {};

  // A staging: thread t covers row t/2, 32B chunk (t&1) of the 64B row.
  // B staging: thread t covers the full 64B row (column) t of 256.
  const int ldRow = tid >> 1;
  const int ldOff = (tid & 1) * 32;
  const char* wSrc = (const char*)(W + (size_t)(mBlk + ldRow) * K) + ldOff;
  const char* dSrc = (const char*)(D + (size_t)(nBlk + tid) * K);
  char* aDst = lds + ldRow * LSTR + ldOff;
  char* bDst = lds + ABYTES + tid * LSTR;

  auto stage = [&](int k0, int buf) {
#if HAVE_ASYNC_LDS
    async_cp16(wSrc + (size_t)k0 * 2,      aDst + buf * BUFB);
    async_cp16(wSrc + (size_t)k0 * 2 + 16, aDst + buf * BUFB + 16);
    async_cp16(dSrc + (size_t)k0 * 2,      bDst + buf * BUFB);
    async_cp16(dSrc + (size_t)k0 * 2 + 16, bDst + buf * BUFB + 16);
    async_cp16(dSrc + (size_t)k0 * 2 + 32, bDst + buf * BUFB + 32);
    async_cp16(dSrc + (size_t)k0 * 2 + 48, bDst + buf * BUFB + 48);
#else
    const uint4* sa = (const uint4*)(wSrc + (size_t)k0 * 2);
    uint4 a0 = sa[0], a1 = sa[1];
    const uint4* sb = (const uint4*)(dSrc + (size_t)k0 * 2);
    uint4 b0v = sb[0], b1v = sb[1], b2v = sb[2], b3v = sb[3];
    uint4* da = (uint4*)(aDst + buf * BUFB);
    da[0] = a0; da[1] = a1;
    uint4* db = (uint4*)(bDst + buf * BUFB);
    db[0] = b0v; db[1] = b1v; db[2] = b2v; db[3] = b3v;
    if (k0 + 32 < K) {
      __builtin_prefetch(wSrc + (size_t)(k0 + 32) * 2, 0, 3);
      __builtin_prefetch(dSrc + (size_t)(k0 + 32) * 2, 0, 3);
    }
#endif
  };

  auto compute = [&](int buf) {
    const char* A  = lds + buf * BUFB;
    const char* Bp = A + ABYTES;

    // B fragments: column (n) contiguous 64B in LDS; lane<16 -> K0..15,
    // lane>=16 -> K16..31 (contiguous k order, same as A).
    bf16x16 bfr[4];
#pragma unroll
    for (int sn = 0; sn < 4; ++sn) {
      const char* p = Bp + (wn * 64 + sn * 16 + l15) * LSTR + hi * 32;
      union { uint4 u[2]; bf16x16 v; } t;
      t.u[0] = *(const uint4*)(p);
      t.u[1] = *(const uint4*)(p + 16);
      bfr[sn] = t.v;
    }
    // A fragments: row (m) per lane; lane<16 -> K0..15 at bytes
    // {0..31}; lane>=16 -> K16..31 at {32..63} (matches B).
    bf16x16 afr[4];
#pragma unroll
    for (int sm = 0; sm < 4; ++sm) {
      const char* p = A + (wm * 64 + sm * 16 + l15) * LSTR + hi * 32;
      union { uint4 u[2]; bf16x16 v; } t;
      t.u[0] = *(const uint4*)(p);
      t.u[1] = *(const uint4*)(p + 16);
      afr[sm] = t.v;
    }
#pragma unroll
    for (int sm = 0; sm < 4; ++sm)
#pragma unroll
      for (int sn = 0; sn < 4; ++sn)
        acc[sm][sn] = wmma_bf16(afr[sm], bfr[sn], acc[sm][sn]);
  };

  const int steps = K >> 5;
  stage(0, 0);
  // steady state: stage tile s+1, drain tile s, compute tile s
  for (int s = 0; s < steps - 1; ++s) {
    const int buf = s & 1;
    stage((s + 1) << 5, buf ^ 1);
#if HAVE_ASYNC_LDS
    wait_async<6>();
#endif
    __syncthreads();   // all waves' tile-s data resident in LDS
    compute(buf);
    __syncthreads();   // done reading tile s; its buffer may be overwritten
  }
#if HAVE_ASYNC_LDS
  wait_async<0>();
#endif
  __syncthreads();
  compute((steps - 1) & 1);

  // Epilogue. C layout: VGPR r -> M = r (lanes 0-15) / 8+r (lanes 16-31),
  // N = lane%16. Each lane stores 8 consecutive o's as one b128.
#pragma unroll
  for (int sm = 0; sm < 4; ++sm) {
#pragma unroll
    for (int sn = 0; sn < 4; ++sn) {
      const int o = mBlk + wm * 64 + sm * 16 + hi * 8;
      const int n = nBlk + wn * 64 + sn * 16 + l15;
      floatx8 c = acc[sm][sn];
      if constexpr (BIAS) {
#pragma unroll
        for (int r = 0; r < 8; ++r) c[r] += bias[o + r];
      }
      if constexpr (RELU) {
#pragma unroll
        for (int r = 0; r < 8; ++r) c[r] = fmaxf(c[r], 0.f);
      }
      if constexpr (MASK) {
        const int bb = n / nPerB;
        const bf16_t* mk = mask + (size_t)bb * maskStride + o;
#pragma unroll
        for (int r = 0; r < 8; ++r)
          if (!((float)mk[r] > 0.f)) c[r] = 0.f;
      }
      PackBF8 st;
#pragma unroll
      for (int r = 0; r < 8; ++r) st.h[r] = (bf16_t)c[r];
      *(uint4*)(Out + (size_t)n * outStride + o) = st.u;
    }
  }
}

// ---------------------------------------------------------------------------
// 6. Head mask array: rows 0..23 = pre-relu a_ld (sign -> gp), else 1.0.
// ---------------------------------------------------------------------------
__global__ void hact_kernel(const bf16_t* __restrict__ lhead, bf16_t* __restrict__ hact)
{
  int idx = blockIdx.x * 256 + threadIdx.x;  // exact grid: BATCH*HM
  int row = idx % HM;
  hact[idx] = (row < N_DOF) ? lhead[idx] : (bf16_t)1.0f;
}

// ---------------------------------------------------------------------------
// 7. Residual MLP 512->8->4->1.
// ---------------------------------------------------------------------------
__global__ void residual_kernel(const bf16_t* __restrict__ h1b,
                                const float* __restrict__ rW0, const float* __restrict__ rb0,
                                const float* __restrict__ rW1, const float* __restrict__ rb1,
                                const float* __restrict__ rW2, const float* __restrict__ rb2,
                                float* __restrict__ res)
{
  int b = blockIdx.x * 256 + threadIdx.x;
  if (b >= BATCH) return;
  const bf16_t* h = h1b + (size_t)b * WIDTH;
  float r0[8];
#pragma unroll
  for (int o = 0; o < 8; ++o) {
    float a = rb0[o];
    const float* wrow = rW0 + o * WIDTH;
    for (int i = 0; i < WIDTH; ++i) a += (float)h[i] * wrow[i];
    r0[o] = fmaxf(a, 0.f);
  }
  float r1[4];
#pragma unroll
  for (int o = 0; o < 4; ++o) {
    float a = rb1[o];
#pragma unroll
    for (int i = 0; i < 8; ++i) a += r0[i] * rW1[o * 8 + i];
    r1[o] = fmaxf(a, 0.f);
  }
  float s = rb2[0];
#pragma unroll
  for (int i = 0; i < 4; ++i) s += r1[i] * rW2[i];
  res[b] = s;
}

// ---------------------------------------------------------------------------
// 8. der0[bn][o] = (y0[b,o] > 0) * W0[o, n]   (bn = b*24+n)
// ---------------------------------------------------------------------------
__global__ void der0_kernel(const bf16_t* __restrict__ y0b, const float* __restrict__ W0,
                            bf16_t* __restrict__ der0)
{
  size_t idx = (size_t)blockIdx.x * 256 + threadIdx.x;  // exact: BN*64
  int chunk = (int)(idx & 63);
  size_t bn = idx >> 6;
  int b = (int)(bn / N_DOF);
  int n = (int)(bn % N_DOF);
  int i0 = chunk * 8;
  const bf16_t* y = y0b + (size_t)b * WIDTH;
  PackBF8 st;
#pragma unroll
  for (int j = 0; j < 8; ++j) {
    int i = i0 + j;
    st.h[j] = ((float)y[i] > 0.f) ? (bf16_t)W0[i * N_DOF + n] : (bf16_t)0.f;
  }
  *(uint4*)(der0 + bn * WIDTH + i0) = st.u;
}

// ---------------------------------------------------------------------------
// 12. Per-batch assembly (one wave32 per batch element).
// quad_dq[k] = 2 qd^T Ldq[k] (L^T qd);  Hdt qd = L(Ldt^T qd)+Ldt(L^T qd).
// ---------------------------------------------------------------------------
__global__ __launch_bounds__(32)
void assemble_kernel(const bf16_t* __restrict__ derH, const bf16_t* __restrict__ lhead,
                     const float* __restrict__ qd, const float* __restrict__ qdd,
                     const float* __restrict__ res, float* __restrict__ out)
{
  const int b = blockIdx.x;
  const int tid = threadIdx.x;
  __shared__ float Lm[N_DOF][N_DOF + 1];
  __shared__ float Ldt[N_DOF][N_DOF + 1];
  __shared__ float derl[M_TRIL][N_DOF];
  __shared__ float qds[N_DOF], qdds[N_DOF], gv[N_DOF];
  __shared__ float u[N_DOF], w[N_DOF], t1[N_DOF];
  __shared__ float Hqdd[N_DOF], Hdtqd[N_DOF];

  for (int i = tid; i < N_DOF * (N_DOF + 1); i += 32) {
    ((float*)Lm)[i] = 0.f; ((float*)Ldt)[i] = 0.f;
  }
  if (tid < N_DOF) {
    qds[tid]  = qd[b * N_DOF + tid];
    qdds[tid] = qdd[b * N_DOF + tid];
    gv[tid]   = (float)derH[((size_t)b * N_DOF + tid) * HM + M_TRIL];
  }
  __syncthreads();

  for (int m = tid; m < M_TRIL; m += 32) {
    int r = (int)((sqrtf(8.f * m + 1.f) - 1.f) * 0.5f);
    while ((r + 1) * (r + 2) / 2 <= m) ++r;
    while (r * (r + 1) / 2 > m) --r;
    int c = m - r * (r + 1) / 2;
    int row = (r == c) ? r : (N_DOF + (m - r));   // diag head row r, else lower
    float a = (float)lhead[(size_t)b * HM + row];
    Lm[r][c] = (r == c) ? fmaxf(a, 0.f) : a;
    float dot = 0.f;
    for (int n = 0; n < N_DOF; ++n) {
      float d = (float)derH[((size_t)b * N_DOF + n) * HM + row];
      derl[m][n] = d;
      dot += d * qds[n];
    }
    Ldt[r][c] = dot;
  }
  __syncthreads();

  if (tid < N_DOF) {
    const int j = tid;
    float su = 0.f, st = 0.f, sw = 0.f;
    for (int i = 0; i < N_DOF; ++i) {
      su += Lm[i][j] * qds[i];
      st += Lm[i][j] * qdds[i];
      sw += Ldt[i][j] * qds[i];
    }
    u[j] = su; t1[j] = st; w[j] = sw;
  }
  __syncthreads();

  if (tid < N_DOF) {
    const int i = tid;
    float hq = 0.f, hd = 0.f;
    for (int j = 0; j < N_DOF; ++j) {
      hq += Lm[i][j] * t1[j];
      hd += Lm[i][j] * w[j] + Ldt[i][j] * u[j];
    }
    Hqdd[i]  = hq + EPSV * qdds[i];
    Hdtqd[i] = hd;
  }
  __syncthreads();

  if (tid < N_DOF) {
    const int k = tid;
    float s = 0.f;
    int m = 0;
    for (int r = 0; r < N_DOF; ++r)
      for (int c = 0; c <= r; ++c) {
        s += derl[m][k] * qds[r] * u[c];
        ++m;
      }
    float quad = 2.f * s;
    out[(size_t)b * N_DOF + k] =
        Hqdd[k] + (Hdtqd[k] - 0.5f * quad) + gv[k] + res[b];
  }
  __syncthreads();

  if (tid == 0) {
    float p = 0.f;
    for (int i = 0; i < N_DOF; ++i)
      p += qds[i] * Hqdd[i] + 0.5f * qds[i] * Hdtqd[i] + qds[i] * gv[i];
    out[(size_t)BATCH * N_DOF + b] = p;
  }
}

// ---------------------------------------------------------------------------
extern "C" void kernel_launch(void* const* d_in, const int* in_sizes, int n_in,
                              void* d_out, int out_size, void* d_ws, size_t ws_size,
                              hipStream_t stream)
{
  (void)in_sizes; (void)n_in; (void)out_size; (void)ws_size;
  const float* q   = (const float*)d_in[0];
  const float* qd  = (const float*)d_in[1];
  const float* qdd = (const float*)d_in[2];
  const float* W0  = (const float*)d_in[3];
  const float* b0  = (const float*)d_in[4];
  const float* W1  = (const float*)d_in[5];
  const float* b1  = (const float*)d_in[6];
  const float* W2  = (const float*)d_in[7];
  const float* b2  = (const float*)d_in[8];
  const float* rW0 = (const float*)d_in[9];
  const float* rb0 = (const float*)d_in[10];
  const float* rW1 = (const float*)d_in[11];
  const float* rb1 = (const float*)d_in[12];
  const float* rW2 = (const float*)d_in[13];
  const float* rb2 = (const float*)d_in[14];
  const float* Wg  = (const float*)d_in[15];
  const float* bg  = (const float*)d_in[16];
  const float* Wlo = (const float*)d_in[17];
  const float* blo = (const float*)d_in[18];
  const float* Wld = (const float*)d_in[19];
  const float* bld = (const float*)d_in[20];
  float* out = (float*)d_out;

  // workspace carve-out (256B aligned)
  char* p = (char*)d_ws;
  auto take = [&](size_t bytes) -> char* {
    char* r = p;
    p += (bytes + 255) & ~(size_t)255;
    return r;
  };
  bf16_t* W1b   = (bf16_t*)take((size_t)WIDTH * WIDTH * 2);
  bf16_t* W2b   = (bf16_t*)take((size_t)WIDTH * WIDTH * 2);
  bf16_t* Wcat  = (bf16_t*)take((size_t)HM * WIDTH * 2);
  float*  bcat  = (float*) take((size_t)HM * 4);
  bf16_t* y0b   = (bf16_t*)take((size_t)BATCH * WIDTH * 2);
  bf16_t* h1b   = (bf16_t*)take((size_t)BATCH * WIDTH * 2);
  bf16_t* y2b   = (bf16_t*)take((size_t)BATCH * WIDTH * 2);
  bf16_t* lhead = (bf16_t*)take((size_t)BATCH * HM * 2);
  bf16_t* hact  = (bf16_t*)take((size_t)BATCH * HM * 2);
  float*  resb  = (float*) take((size_t)BATCH * 4);
  bf16_t* bufA  = (bf16_t*)take((size_t)BN * WIDTH * 2);  // der0, reused as derH
  bf16_t* der1  = (bf16_t*)take((size_t)BN * WIDTH * 2);
  bf16_t* der2  = (bf16_t*)take((size_t)BN * WIDTH * 2);
  bf16_t* der0  = bufA;
  bf16_t* derH  = bufA;  // der0 is dead once der1 GEMM finishes

  // 1. prep
  {
    int total = WIDTH * WIDTH * 2 + HM * WIDTH + HM;
    prep_kernel<<<(total + 255) / 256, 256, 0, stream>>>(
        W1, W2, Wld, Wlo, Wg, bld, blo, bg, W1b, W2b, Wcat, bcat);
  }
  // 2. layer0
  layer0_kernel<<<(BATCH * WIDTH) / 256, 256, 0, stream>>>(q, W0, b0, y0b);
  // 3. h1  4. y2  5. lhead
  gemm_bf16_wmma<true, true, false><<<dim3(BATCH / 256, WIDTH / 128), 256, 0, stream>>>(
      W1b, y0b, h1b, b1, nullptr, WIDTH, 1, 0, WIDTH);
  gemm_bf16_wmma<true, true, false><<<dim3(BATCH / 256, WIDTH / 128), 256, 0, stream>>>(
      W2b, h1b, y2b, b2, nullptr, WIDTH, 1, 0, WIDTH);
  gemm_bf16_wmma<true, false, false><<<dim3(BATCH / 256, HM / 128), 256, 0, stream>>>(
      Wcat, y2b, lhead, bcat, nullptr, WIDTH, 1, 0, HM);
  // 6. hact  7. residual  8. der0
  hact_kernel<<<(BATCH * HM) / 256, 256, 0, stream>>>(lhead, hact);
  residual_kernel<<<BATCH / 256, 256, 0, stream>>>(h1b, rW0, rb0, rW1, rb1, rW2, rb2, resb);
  der0_kernel<<<((size_t)BN * 64) / 256, 256, 0, stream>>>(y0b, W0, der0);
  // 9-11. Jacobian GEMMs (the 70 GMAC core)
  gemm_bf16_wmma<false, false, true><<<dim3(BN / 256, WIDTH / 128), 256, 0, stream>>>(
      W1b, der0, der1, nullptr, h1b, WIDTH, N_DOF, WIDTH, WIDTH);
  gemm_bf16_wmma<false, false, true><<<dim3(BN / 256, WIDTH / 128), 256, 0, stream>>>(
      W2b, der1, der2, nullptr, y2b, WIDTH, N_DOF, WIDTH, WIDTH);
  gemm_bf16_wmma<false, false, true><<<dim3(BN / 256, HM / 128), 256, 0, stream>>>(
      Wcat, der2, derH, nullptr, hact, WIDTH, N_DOF, HM, HM);
  // 12. assemble outputs
  assemble_kernel<<<BATCH, 32, 0, stream>>>(derH, lhead, qd, qdd, resb, out);
}
